// Policy_29824252903773
// MI455X (gfx1250) — matrix operand order repaired and checked by reading, not compiled, not run on hardware
//
#include <hip/hip_runtime.h>
#include <stdint.h>

// ---- CDNA5 WMMA vector types (wave32) ----
typedef __attribute__((ext_vector_type(16))) __bf16 v16bf;
typedef __attribute__((ext_vector_type(8)))  float  v8f;
typedef __attribute__((ext_vector_type(4)))  float  v4f;

#define NBATCH 64
#define POMO   1000
#define EMB    128
#define NN     1000
#define NTILES 63          // 63 * 16 = 1008 >= 1000
#define NPAD   1008
#define SSTRIDE 1020       // LDS row stride in floats; 8*1020 % 64 == 32 -> conflict-free
#define MROWS  64          // P-rows per workgroup
#define MCHUNKS 16         // ceil(1000/64)

__device__ __forceinline__ __bf16 f2bf(float f) {
    // native convert: backend selects v_cvt_pk_bf16_f32
    return (__bf16)f;
}

__global__ __launch_bounds__(256, 1)
void policy_probs_kernel(const float* __restrict__ A,    // [B, POMO, EMB]
                         const float* __restrict__ Kx,   // [B, EMB, NN]
                         const float* __restrict__ Mk,   // [B, POMO, NN]
                         float* __restrict__ Out)        // [B, POMO, NN]
{
    __shared__ float sc[MROWS * SSTRIDE];

    const int b    = blockIdx.x >> 4;        // 64 batches
    const int mc   = blockIdx.x & 15;        // 16 row-chunks
    const int p0   = mc * MROWS;
    const int tid  = threadIdx.x;
    const int wave = tid >> 5;
    const int lane = tid & 31;
    const int lhi  = lane >> 4;              // 0 or 1
    const int llo  = lane & 15;

    const float* Ab = A  + (size_t)b * POMO * EMB;
    const float* Kb = Kx + (size_t)b * EMB  * NN;
    const float* Mb = Mk + (size_t)b * POMO * NN;

    // ---------- load A fragments: 4 M-subtiles x 4 K-steps, bf16 ----------
    // 16-bit A 16x32 layout: lane m = llo, K = lhi*8 + (j<8 ? j : j+8)
    v16bf afrag[4][4];
#pragma unroll
    for (int mt = 0; mt < 4; ++mt) {
        int p = p0 + mt * 16 + llo;
        if (p > POMO - 1) p = POMO - 1;      // clamp (duplicates harmless, never stored)
        const float* ap = Ab + (size_t)p * EMB;
#pragma unroll
        for (int kf = 0; kf < 4; ++kf) {
            const int kb = kf * 32 + lhi * 8;
#pragma unroll
            for (int j = 0; j < 16; ++j) {
                const int k = kb + j + (j >= 8 ? 8 : 0);
                afrag[mt][kf][j] = f2bf(ap[k]);
            }
        }
    }

    const float kscale = 0.17677669529663689f; // 2 / sqrt(128), folded for tanh-via-exp

    // ---------- GEMM + fused epilogue over this wave's N tiles ----------
#pragma unroll 1
    for (int nt = wave; nt < NTILES; nt += 8) {
        const int n  = nt * 16 + llo;
        const int nc = (n < NN) ? n : (NN - 1);

        v8f z = {0.f,0.f,0.f,0.f,0.f,0.f,0.f,0.f};
        v8f acc0 = z, acc1 = z, acc2 = z, acc3 = z;

#pragma unroll
        for (int kf = 0; kf < 4; ++kf) {
            // 16-bit B 32x16 layout: N = llo, K = kf*32 + lhi*8 + (j<8 ? j : j+8)
            const int kb = kf * 32 + lhi * 8;
            v16bf bfrag;
#pragma unroll
            for (int j = 0; j < 16; ++j)
                bfrag[j] = f2bf(Kb[(size_t)(kb + j + (j >= 8 ? 8 : 0)) * NN + nc]);   // temporal: reused across 16 WGs via L2

            acc0 = __builtin_amdgcn_wmma_f32_16x16x32_bf16(false, afrag[0][kf], false, bfrag, (short)0, acc0, false, false);
            acc1 = __builtin_amdgcn_wmma_f32_16x16x32_bf16(false, afrag[1][kf], false, bfrag, (short)0, acc1, false, false);
            acc2 = __builtin_amdgcn_wmma_f32_16x16x32_bf16(false, afrag[2][kf], false, bfrag, (short)0, acc2, false, false);
            acc3 = __builtin_amdgcn_wmma_f32_16x16x32_bf16(false, afrag[3][kf], false, bfrag, (short)0, acc3, false, false);
        }

        // epilogue: scale, 10*tanh, +mask (streaming, NT), stage into LDS
        v8f accs[4] = {acc0, acc1, acc2, acc3};
#pragma unroll
        for (int mt = 0; mt < 4; ++mt) {
#pragma unroll
            for (int r = 0; r < 8; ++r) {
                const int m = mt * 16 + lhi * 8 + r;           // C/D layout: M = r + 8*lhi
                int p = p0 + m;
                const int pc = (p < POMO) ? p : (POMO - 1);
                float t2 = accs[mt][r] * kscale;               // 2*score/sqrt(128)
                t2 = fminf(fmaxf(t2, -30.0f), 30.0f);
                const float e = __expf(t2);
                float val = 10.0f * (e - 1.0f) / (e + 1.0f);   // 10*tanh
                val += __builtin_nontemporal_load(&Mb[(size_t)pc * NN + nc]);
                if (n >= NN) val = -1e30f;                     // pad cols vanish in softmax
                sc[m * SSTRIDE + n] = val;
            }
        }
    }

    __syncthreads();

    // ---------- softmax over N, 8 rows per wave, 128b vectorized ----------
#pragma unroll 1
    for (int r8 = 0; r8 < 8; ++r8) {
        const int row = wave * 8 + r8;
        float* srow = &sc[row * SSTRIDE];          // 4080 B row stride -> 16B aligned
        v4f* srow4 = (v4f*)srow;

        float mx = -1e30f;
#pragma unroll 1
        for (int i = lane; i < NPAD / 4; i += 32) {            // 252 vectors
            const v4f v = srow4[i];
            mx = fmaxf(mx, fmaxf(fmaxf(v.x, v.y), fmaxf(v.z, v.w)));
        }
#pragma unroll
        for (int off = 16; off > 0; off >>= 1)
            mx = fmaxf(mx, __shfl_xor(mx, off, 32));

        float sum = 0.0f;
#pragma unroll 1
        for (int i = lane; i < NPAD / 4; i += 32) {
            v4f v = srow4[i];
            v.x = __expf(v.x - mx);
            v.y = __expf(v.y - mx);
            v.z = __expf(v.z - mx);
            v.w = __expf(v.w - mx);
            srow4[i] = v;
            sum += (v.x + v.y) + (v.z + v.w);
        }
#pragma unroll
        for (int off = 16; off > 0; off >>= 1)
            sum += __shfl_xor(sum, off, 32);

        const float inv = 1.0f / sum;
        const int p = p0 + row;
        if (p < POMO) {
            v4f* op4 = (v4f*)(Out + (size_t)b * POMO * NN + (size_t)p * NN);
#pragma unroll 1
            for (int i = lane; i < NN / 4; i += 32) {          // 250 vectors, 1000 = 4*250
                v4f v = srow4[i];
                v *= inv;
                __builtin_nontemporal_store(v, &op4[i]);       // streaming b128 store, TH=NT
            }
        }
    }
}

extern "C" void kernel_launch(void* const* d_in, const int* in_sizes, int n_in,
                              void* d_out, int out_size, void* d_ws, size_t ws_size,
                              hipStream_t stream) {
    (void)in_sizes; (void)n_in; (void)d_ws; (void)ws_size; (void)out_size;
    const float* A  = (const float*)d_in[0];   // mh_attn_out  [64,1000,128]
    const float* Kx = (const float*)d_in[1];   // single_head_key [64,128,1000]
    const float* Mk = (const float*)d_in[2];   // mask [64,1000,1000]
    float* Out = (float*)d_out;                // probs [64,1000,1000]

    dim3 grid(NBATCH * MCHUNKS);               // 1024 workgroups
    dim3 block(256);                           // 8 wave32 waves
    policy_probs_kernel<<<grid, block, 0, stream>>>(A, Kx, Mk, Out);
}
